// AttentionLayerO2TwoUpdateNodeGeneral_75831942578380
// MI455X (gfx1250) — matrix operand order repaired and checked by reading, not compiled, not run on hardware
//
#include <hip/hip_runtime.h>
#include <hip/hip_bf16.h>
#include <math.h>

#define NNODES 10000
#define NEDGE  240000
#define HID    256
#define HEADS  16
#define HD     16
#define EFK    96   // edge-feature dim padded 84 -> 96 (multiple of 32)

typedef __attribute__((ext_vector_type(16))) __bf16 v16bf;
typedef __attribute__((ext_vector_type(8)))  float  v8f;

// ---------------------------------------------------------------------------
// helpers
// ---------------------------------------------------------------------------
__device__ __forceinline__ void atomic_max_float(float* addr, float val) {
  if (val >= 0.0f) atomicMax((int*)addr, __float_as_int(val));
  else             atomicMin((unsigned int*)addr, __float_as_uint(val));
}

__device__ __forceinline__ float wave_sum(float x) {
#pragma unroll
  for (int o = 16; o > 0; o >>= 1) x += __shfl_xor(x, o, 32);
  return x;
}

__device__ __forceinline__ unsigned int pack_bf16(float lo, float hi) {
  union { __bf16 b; unsigned short u; } a, c;
  a.b = (__bf16)lo; c.b = (__bf16)hi;
  return ((unsigned int)c.u << 16) | (unsigned int)a.u;
}

union FragU { unsigned int u[8]; v16bf v; };

// ---------------------------------------------------------------------------
// WMMA GEMM: Y[M,N] = act(X[M,K] @ W[K,N] + bias), f32 in/out, bf16 WMMA.
// REQUIRES: K % 32 == 0 (true here: K in {96,256,512}), N % 64 == 0,
// 16B-aligned rows. Block = 256 threads = 8 waves; block tile 128x64;
// wave tile 32x32 (4 WMMA / K-step, A & B fragments each reused twice).
// Double-buffered LDS: one barrier per K-step, staging overlaps WMMA.
// ACT: 0 = none, 1 = tanh, 2 = silu
// ---------------------------------------------------------------------------
template<int ACT>
__global__ __launch_bounds__(256)
void gemm_kernel(const float* __restrict__ X, const float* __restrict__ W,
                 const float* __restrict__ bias, float* __restrict__ Y,
                 int M, int K, int N) {
  // bf16 tiles. Row stride 20 dwords (80B): 16B-aligned rows, 16 consecutive
  // rows map to distinct banks for ds_load_b128.
  __shared__ unsigned int As[2][128][20];  // A tile 128x32, row-major bf16 pairs
  __shared__ unsigned int Bt[2][64][20];   // B tile 32x64, transposed: Bt[n][k]

  const int tid  = threadIdx.x;
  const int lane = tid & 31;
  const int wave = tid >> 5;
  const int wr   = wave >> 1;              // 0..3 : 32-row slice
  const int wc   = wave & 1;               // 0..1 : 32-col slice
  const int rowBlk = blockIdx.x * 128;
  const int colBlk = blockIdx.y * 64;

  v8f acc[2][2];
#pragma unroll
  for (int i = 0; i < 2; i++)
#pragma unroll
    for (int j = 0; j < 2; j++)
      acc[i][j] = (v8f){0.f,0.f,0.f,0.f,0.f,0.f,0.f,0.f};

  const int lr   = lane & 15;
  const int hi   = (lane >> 4) & 1;
  const int aOff = hi ? 4 : 0;             // A frag: bf16 K-base {0,8}  -> dword {0,4}
  const int bOff = hi ? 4 : 0;             // B frag: bf16 K-base {0,8}  -> dword {0,4}

  // branchless stage of one K-step tile into buffer `buf`
  auto stage = [&](int kb, int buf) {
    // A: 128 rows x 32 cols, 4 float4 per thread, row-clamped (no zeroing
    // needed: out-of-range rows are never stored)
#pragma unroll
    for (int i = 0; i < 4; i++) {
      int idx = tid + i * 256;             // 0..1023
      int r   = idx >> 3;                  // 0..127
      int c4  = (idx & 7) << 2;            // 0,4,..,28
      int gr  = rowBlk + r; if (gr > M - 1) gr = M - 1;
      const float4 f = *(const float4*)(X + (size_t)gr * K + kb + c4);
      As[buf][r][(c4 >> 1) + 0] = pack_bf16(f.x, f.y);
      As[buf][r][(c4 >> 1) + 1] = pack_bf16(f.z, f.w);
    }
    // B: 32 k-rows x 64 cols -> transposed bf16 Bt[n][k]; two consecutive
    // k-rows per thread so k-pairs pack into single dword stores
    {
      int k2 = (tid >> 4) << 1;            // 0,2,..,30
      int c4 = (tid & 15) << 2;            // n: 0,4,..,60
      const float* wp = W + (size_t)(kb + k2) * N + colBlk + c4;
      const float4 f0 = *(const float4*)(wp);
      const float4 f1 = *(const float4*)(wp + N);
      int kd = k2 >> 1;                    // dword col 0..15
      Bt[buf][c4 + 0][kd] = pack_bf16(f0.x, f1.x);
      Bt[buf][c4 + 1][kd] = pack_bf16(f0.y, f1.y);
      Bt[buf][c4 + 2][kd] = pack_bf16(f0.z, f1.z);
      Bt[buf][c4 + 3][kd] = pack_bf16(f0.w, f1.w);
    }
  };

  const int nk = K >> 5;
  stage(0, 0);
  __syncthreads();

  for (int t = 0; t < nk; t++) {
    const int buf = t & 1;
    if (t + 1 < nk) {
      stage((t + 1) << 5, buf ^ 1);        // overlap next-tile staging with WMMA
      if (t + 2 < nk) {
        int pr = rowBlk + (tid >> 1); if (pr > M - 1) pr = M - 1;
        __builtin_prefetch(X + (size_t)pr * K + ((t + 2) << 5));
        __builtin_prefetch(W + (size_t)(((t + 2) << 5) + lane) * N + colBlk);
      }
    }

    // fragments: 2x ds_load_b128 each, A & B each reused for 2 WMMAs
    FragU a[2], b[2];
#pragma unroll
    for (int i = 0; i < 2; i++) {
      const int arow = wr * 32 + i * 16 + lr;
      uint4 p = *(const uint4*)&As[buf][arow][aOff];      // bf16 K {base..base+7}
      uint4 q = *(const uint4*)&As[buf][arow][aOff + 8];  // bf16 K {base+16..base+23}
      a[i].u[0]=p.x; a[i].u[1]=p.y; a[i].u[2]=p.z; a[i].u[3]=p.w;
      a[i].u[4]=q.x; a[i].u[5]=q.y; a[i].u[6]=q.z; a[i].u[7]=q.w;
    }
#pragma unroll
    for (int j = 0; j < 2; j++) {
      const int bcol = wc * 32 + j * 16 + lr;
      uint4 p = *(const uint4*)&Bt[buf][bcol][bOff];      // bf16 K {base..base+7}
      uint4 q = *(const uint4*)&Bt[buf][bcol][bOff + 8];  // bf16 K {base+16..base+23}
      b[j].u[0]=p.x; b[j].u[1]=p.y; b[j].u[2]=p.z; b[j].u[3]=p.w;
      b[j].u[4]=q.x; b[j].u[5]=q.y; b[j].u[6]=q.z; b[j].u[7]=q.w;
    }
#pragma unroll
    for (int i = 0; i < 2; i++)
#pragma unroll
      for (int j = 0; j < 2; j++)
        acc[i][j] = __builtin_amdgcn_wmma_f32_16x16x32_bf16(
            false, a[i].v, false, b[j].v, (short)0, acc[i][j], false, false);
    __syncthreads();
  }

  const int col0 = colBlk + wc * 32 + lr;
  const int col1 = col0 + 16;
  const float bv0 = bias ? bias[col0] : 0.0f;
  const float bv1 = bias ? bias[col1] : 0.0f;
#pragma unroll
  for (int i = 0; i < 2; i++) {
    const int rbase = rowBlk + wr * 32 + i * 16 + hi * 8;
#pragma unroll
    for (int r = 0; r < 8; r++) {
      int row = rbase + r;
      if (row < M) {
        float y0 = acc[i][0][r] + bv0;
        float y1 = acc[i][1][r] + bv1;
        if (ACT == 1)      { y0 = tanhf(y0); y1 = tanhf(y1); }
        else if (ACT == 2) { y0 = y0 / (1.0f + __expf(-y0)); y1 = y1 / (1.0f + __expf(-y1)); }
        Y[(size_t)row * N + col0] = y0;
        Y[(size_t)row * N + col1] = y1;
      }
    }
  }
}

// ---------------------------------------------------------------------------
// edge features: dist -> 20 gaussians x 4 edge types + 4 raw, padded to 96
// ---------------------------------------------------------------------------
__global__ void edge_feat_kernel(const float* __restrict__ pos,
                                 const float* __restrict__ etype,
                                 const int* __restrict__ eidx,
                                 float* __restrict__ ef, int E) {
  int e = blockIdx.x * blockDim.x + threadIdx.x;
  if (e >= E) return;
  int s = eidx[e], d = eidx[E + e];
  float dx = pos[d * 3 + 0] - pos[s * 3 + 0];
  float dy = pos[d * 3 + 1] - pos[s * 3 + 1];
  float dz = pos[d * 3 + 2] - pos[s * 3 + 2];
  float dist = sqrtf(dx * dx + dy * dy + dz * dz + 1e-12f);
  const float step  = 10.0f / 19.0f;
  const float coeff = -0.5f / (step * step);
  float g[20];
#pragma unroll
  for (int i = 0; i < 20; i++) {
    float df = dist - step * (float)i;
    g[i] = __expf(coeff * df * df);
  }
  float* o = ef + (size_t)e * EFK;
#pragma unroll
  for (int t = 0; t < 4; t++) {
    float et = etype[e * 4 + t];
#pragma unroll
    for (int i = 0; i < 20; i++) o[t * 20 + i] = et * g[i];
    o[80 + t] = et;
  }
#pragma unroll
  for (int i = 84; i < EFK; i++) o[i] = 0.0f;
}

// ---------------------------------------------------------------------------
// attention scatter/gather kernels
// ---------------------------------------------------------------------------
__global__ void attn_init_kernel(float* m, float* s, float* agg, int nmh, int nagg) {
  int t = blockIdx.x * blockDim.x + threadIdx.x;
  if (t < nmh) { m[t] = -3.0e38f; s[t] = 0.0f; }
  if (t < nagg) agg[t] = 0.0f;
}

__global__ void attn_logits_kernel(const float* __restrict__ q, const float* __restrict__ k,
                                   const float* __restrict__ e0,
                                   const int* __restrict__ srcI, const int* __restrict__ dstI,
                                   float* __restrict__ alpha, float* __restrict__ mbuf, int E) {
  int t = blockIdx.x * blockDim.x + threadIdx.x;
  if (t >= E * HEADS) return;
  int e = t >> 4, hh = t & 15;
  int s = srcI[e], d = dstI[e];
  const float* qp = q  + (size_t)d * HID + hh * HD;
  const float* kp = k  + (size_t)s * HID + hh * HD;
  const float* ep = e0 + (size_t)e * HID + hh * HD;
  float acc = 0.0f;
#pragma unroll
  for (int c = 0; c < HD; c++) acc += qp[c] * kp[c] * ep[c];
  acc *= 0.25f;                        // 1/sqrt(HD), HD=16
  alpha[t] = acc;
  atomic_max_float(&mbuf[d * HEADS + hh], acc);
}

__global__ void attn_expsum_kernel(float* __restrict__ alpha, const float* __restrict__ mbuf,
                                   float* __restrict__ sbuf, const int* __restrict__ dstI, int E) {
  int t = blockIdx.x * blockDim.x + threadIdx.x;
  if (t >= E * HEADS) return;
  int e = t >> 4, hh = t & 15;
  int d = dstI[e];
  float a = __expf(alpha[t] - mbuf[d * HEADS + hh]);
  alpha[t] = a;
  atomicAdd(&sbuf[d * HEADS + hh], a);
}

__global__ void attn_message_kernel(const float* __restrict__ v, const float* __restrict__ e1,
                                    const float* __restrict__ alpha, const float* __restrict__ sbuf,
                                    const int* __restrict__ srcI, const int* __restrict__ dstI,
                                    float* __restrict__ agg, int E) {
  int t = blockIdx.x * blockDim.x + threadIdx.x;
  if (t >= E * HID) return;
  int e = t >> 8, j = t & 255;
  int hh = j >> 4;
  int s = srcI[e], d = dstI[e];
  float w = alpha[e * HEADS + hh] / sbuf[d * HEADS + hh];
  atomicAdd(&agg[(size_t)d * HID + j], v[(size_t)s * HID + j] * e1[(size_t)e * HID + j] * w);
}

// ---------------------------------------------------------------------------
// elementwise / rowwise kernels
// ---------------------------------------------------------------------------
__global__ void concat_kernel(const float* __restrict__ a, const float* __restrict__ b,
                              float* __restrict__ c, int M) {
  int t = blockIdx.x * blockDim.x + threadIdx.x;
  if (t >= M * 512) return;
  int r = t >> 9, j = t & 511;
  c[t] = (j < 256) ? a[(size_t)r * 256 + j] : b[(size_t)r * 256 + (j - 256)];
}

__global__ void ln_relu_kernel(float* __restrict__ x, int M) {  // eps 1e-5, relu
  int row  = blockIdx.x * 8 + (threadIdx.x >> 5);
  int lane = threadIdx.x & 31;
  if (row >= M) return;
  float* p = x + (size_t)row * 256;
  float vals[8]; float s = 0.f;
#pragma unroll
  for (int i = 0; i < 8; i++) { vals[i] = p[lane + i * 32]; s += vals[i]; }
  float mu = wave_sum(s) * (1.0f / 256.0f);
  float var = 0.f;
#pragma unroll
  for (int i = 0; i < 8; i++) { float d = vals[i] - mu; var += d * d; }
  var = wave_sum(var) * (1.0f / 256.0f);
  float rstd = rsqrtf(var + 1e-5f);
#pragma unroll
  for (int i = 0; i < 8; i++) p[lane + i * 32] = fmaxf((vals[i] - mu) * rstd, 0.0f);
}

__global__ void residual_ln_kernel(const float* __restrict__ h, const float* __restrict__ y,
                                   float* __restrict__ out, int M) {  // eps 1e-6
  int row  = blockIdx.x * 8 + (threadIdx.x >> 5);
  int lane = threadIdx.x & 31;
  if (row >= M) return;
  const float* hp = h + (size_t)row * 256;
  const float* yp = y + (size_t)row * 256;
  float* op = out + (size_t)row * 256;
  float vals[8]; float s = 0.f;
#pragma unroll
  for (int i = 0; i < 8; i++) { vals[i] = hp[lane + i * 32] + yp[lane + i * 32]; s += vals[i]; }
  float mu = wave_sum(s) * (1.0f / 256.0f);
  float var = 0.f;
#pragma unroll
  for (int i = 0; i < 8; i++) { float d = vals[i] - mu; var += d * d; }
  var = wave_sum(var) * (1.0f / 256.0f);
  float rstd = rsqrtf(var + 1e-6f);
#pragma unroll
  for (int i = 0; i < 8; i++) op[lane + i * 32] = (vals[i] - mu) * rstd;
}

__global__ void add_inplace_kernel(float* __restrict__ out, const float* __restrict__ g, int n) {
  int t = blockIdx.x * blockDim.x + threadIdx.x;
  if (t < n) out[t] += g[t];
}

// ---------------------------------------------------------------------------
// host orchestration
// ---------------------------------------------------------------------------
static inline void launch_gemm(const float* X, const float* W, const float* b, float* Y,
                               int M, int K, int N, int act, hipStream_t st) {
  dim3 grid((M + 127) / 128, N / 64);
  if (act == 0)      gemm_kernel<0><<<grid, 256, 0, st>>>(X, W, b, Y, M, K, N);
  else if (act == 1) gemm_kernel<1><<<grid, 256, 0, st>>>(X, W, b, Y, M, K, N);
  else               gemm_kernel<2><<<grid, 256, 0, st>>>(X, W, b, Y, M, K, N);
}

extern "C" void kernel_launch(void* const* d_in, const int* in_sizes, int n_in,
                              void* d_out, int out_size, void* d_ws, size_t ws_size,
                              hipStream_t stream) {
  (void)in_sizes; (void)n_in; (void)out_size; (void)ws_size;
  const float* h       = (const float*)d_in[0];
  const float* h_bond  = (const float*)d_in[1];
  const float* pos     = (const float*)d_in[2];
  const float* etype   = (const float*)d_in[3];
  const int*   eidx    = (const int*)d_in[4];
  const int*   bidx    = (const int*)d_in[5];
  const float* Wq1     = (const float*)d_in[6];
  const float* Wk1     = (const float*)d_in[7];
  const float* Wv1     = (const float*)d_in[8];
  const float* We0_1   = (const float*)d_in[9];
  const float* We1_1   = (const float*)d_in[10];
  const float* mlp1_W1 = (const float*)d_in[11];
  const float* mlp1_b1 = (const float*)d_in[12];
  const float* mlp1_W2 = (const float*)d_in[13];
  const float* mlp1_b2 = (const float*)d_in[14];
  const float* Wq2     = (const float*)d_in[15];
  const float* Wk2     = (const float*)d_in[16];
  const float* Wv2     = (const float*)d_in[17];
  const float* We0_2   = (const float*)d_in[18];
  const float* We1_2   = (const float*)d_in[19];
  const float* mlp2_W1 = (const float*)d_in[20];
  const float* mlp2_b1 = (const float*)d_in[21];
  const float* mlp2_W2 = (const float*)d_in[22];
  const float* mlp2_b2 = (const float*)d_in[23];
  const float* lin_W   = (const float*)d_in[24];
  const float* lin_b   = (const float*)d_in[25];
  const float* ff1_W   = (const float*)d_in[26];
  const float* ff1_b   = (const float*)d_in[27];
  const float* ff2_W   = (const float*)d_in[28];
  const float* ff2_b   = (const float*)d_in[29];

  float* out = (float*)d_out;
  float* ws  = (float*)d_ws;

  const int N = NNODES, E = NEDGE, EB = NEDGE;
  size_t off = 0;
  auto alloc = [&](size_t n) { float* p = ws + off; off += n; return p; };
  float* ef   = alloc((size_t)E * EFK);     // padded edge features [E,96]
  float* efW0 = alloc((size_t)EFK * HID);   // zero-padded We0_1 [96,256]
  float* efW1 = alloc((size_t)EFK * HID);   // zero-padded We1_1 [96,256]
  float* e0b  = alloc((size_t)E * HID);
  float* e1b  = alloc((size_t)E * HID);
  float* qb   = alloc((size_t)N * HID);
  float* kb   = alloc((size_t)N * HID);
  float* vb   = alloc((size_t)N * HID);
  float* al   = alloc((size_t)E * HEADS);
  float* mb   = alloc((size_t)N * HEADS);
  float* sb   = alloc((size_t)N * HEADS);
  float* agg  = alloc((size_t)N * HID);
  float* cat  = alloc((size_t)N * 512);
  float* t1   = alloc((size_t)N * 512);
  float* t2   = alloc((size_t)N * 512);
  float* h1   = alloc((size_t)N * HID);
  float* h2   = alloc((size_t)N * HID);

  const int thr = 256;
  const int gEH  = (E * HEADS + thr - 1) / thr;
  const int gEC  = (int)(((size_t)E * HID + thr - 1) / thr);
  const int gN   = (N + 7) / 8;
  const int gNC  = (N * HID + thr - 1) / thr;
  const int gCAT = (N * 512 + thr - 1) / thr;

  // zero-pad edge-projection weights 84x256 -> 96x256 (rows 84..95 = 0)
  hipMemcpyAsync(efW0, We0_1, (size_t)84 * HID * sizeof(float), hipMemcpyDeviceToDevice, stream);
  hipMemcpyAsync(efW1, We1_1, (size_t)84 * HID * sizeof(float), hipMemcpyDeviceToDevice, stream);
  hipMemsetAsync(efW0 + (size_t)84 * HID, 0, (size_t)(EFK - 84) * HID * sizeof(float), stream);
  hipMemsetAsync(efW1 + (size_t)84 * HID, 0, (size_t)(EFK - 84) * HID * sizeof(float), stream);

  // ---------------- layer 1: geometric-edge attention ----------------
  edge_feat_kernel<<<(E + thr - 1) / thr, thr, 0, stream>>>(pos, etype, eidx, ef, E);
  launch_gemm(ef, efW0, nullptr, e0b, E, EFK, HID, 1, stream);
  launch_gemm(ef, efW1, nullptr, e1b, E, EFK, HID, 1, stream);
  launch_gemm(h, Wq1, nullptr, qb, N, HID, HID, 0, stream);
  launch_gemm(h, Wk1, nullptr, kb, N, HID, HID, 0, stream);
  launch_gemm(h, Wv1, nullptr, vb, N, HID, HID, 0, stream);
  attn_init_kernel<<<gNC, thr, 0, stream>>>(mb, sb, agg, N * HEADS, N * HID);
  const int* src1 = eidx; const int* dst1 = eidx + E;
  attn_logits_kernel<<<gEH, thr, 0, stream>>>(qb, kb, e0b, src1, dst1, al, mb, E);
  attn_expsum_kernel<<<gEH, thr, 0, stream>>>(al, mb, sb, dst1, E);
  attn_message_kernel<<<gEC, thr, 0, stream>>>(vb, e1b, al, sb, src1, dst1, agg, E);
  concat_kernel<<<gCAT, thr, 0, stream>>>(agg, h, cat, N);
  launch_gemm(cat, mlp1_W1, mlp1_b1, t1, N, 512, HID, 0, stream);
  ln_relu_kernel<<<gN, 256, 0, stream>>>(t1, N);
  launch_gemm(t1, mlp1_W2, mlp1_b2, h1, N, HID, HID, 0, stream);

  // ---------------- layer 2: bond-edge attention ----------------
  launch_gemm(h_bond, We0_2, nullptr, e0b, EB, HID, HID, 1, stream);
  launch_gemm(h_bond, We1_2, nullptr, e1b, EB, HID, HID, 1, stream);
  launch_gemm(h1, Wq2, nullptr, qb, N, HID, HID, 0, stream);
  launch_gemm(h1, Wk2, nullptr, kb, N, HID, HID, 0, stream);
  launch_gemm(h1, Wv2, nullptr, vb, N, HID, HID, 0, stream);
  attn_init_kernel<<<gNC, thr, 0, stream>>>(mb, sb, agg, N * HEADS, N * HID);
  const int* src2 = bidx; const int* dst2 = bidx + EB;
  attn_logits_kernel<<<gEH, thr, 0, stream>>>(qb, kb, e0b, src2, dst2, al, mb, EB);
  attn_expsum_kernel<<<gEH, thr, 0, stream>>>(al, mb, sb, dst2, EB);
  attn_message_kernel<<<gEC, thr, 0, stream>>>(vb, e1b, al, sb, src2, dst2, agg, EB);
  concat_kernel<<<gCAT, thr, 0, stream>>>(agg, h1, cat, N);
  launch_gemm(cat, mlp2_W1, mlp2_b1, t1, N, 512, HID, 0, stream);
  ln_relu_kernel<<<gN, 256, 0, stream>>>(t1, N);
  launch_gemm(t1, mlp2_W2, mlp2_b2, h2, N, HID, HID, 0, stream);

  // ---------------- residual + feed-forward ----------------
  launch_gemm(h2, lin_W, lin_b, t1, N, HID, HID, 0, stream);
  residual_ln_kernel<<<gN, 256, 0, stream>>>(h, t1, out, N);            // hn -> d_out
  launch_gemm(out, ff1_W, ff1_b, t1, N, HID, 512, 2, stream);           // silu
  launch_gemm(t1, ff2_W, ff2_b, t2, N, 512, HID, 0, stream);
  add_inplace_kernel<<<gNC, thr, 0, stream>>>(out, t2, N * HID);
}
